// uni_network_model_31628139167824
// MI455X (gfx1250) — matrix operand order repaired and checked by reading, not compiled, not run on hardware
//
#include <hip/hip_runtime.h>
#include <hip/hip_bf16.h>

// ---------------------------------------------------------------------------
// Spatio-temporal LSTM on MI455X (gfx1250), wave32 + V_WMMA_F32_16X16X32_F16.
// Per timestep: GEMM1 (6400x256 @ 256x256 -> si|sf), GEMM2 (6400x128 @ 128x512
// -> gates), LSTM epilogue, spatial scatter + junction matmuls, output head.
// ---------------------------------------------------------------------------

typedef __attribute__((ext_vector_type(16))) _Float16 v16h;
typedef __attribute__((ext_vector_type(8)))  _Float16 v8h;
typedef __attribute__((ext_vector_type(8)))  float    v8f;
typedef __attribute__((ext_vector_type(4)))  float    v4f;

constexpr int HID  = 128;        // hidden size
constexpr int SPA  = 100;        // spatial
constexpr int BAT  = 64;         // batch
constexpr int TSTP = 128;        // temporal
constexpr int ROWS = BAT * SPA;  // 6400
constexpr int TPRED = 8;
constexpr int TOUT  = TSTP - TPRED; // 120

// Junction buckets (BUCKETS = [15,30,45,60,70,85,99] in the reference)
constexpr int JB1 = 30, JB3 = 60, JB4 = 70, JB6 = 99;

constexpr int MT   = 32;   // rows per block in step_main / head
constexpr int SPLD = 264;  // LDS row stride (f16) for 256-wide sp tile (+8 pad)
constexpr int HLD  = 136;  // LDS row stride (f16) for 128-wide h tile (+8 pad)

__device__ __forceinline__ float sigm(float x) { return 1.f / (1.f + __expf(-x)); }

// A-fragment (16x32 f16, M x K) from LDS, row stride ld (f16 elements).
// lane<16: K = kb+0..7 (v0..3), kb+16..23 (v4..7); lane>=16: +8 on each half.
__device__ __forceinline__ v16h load_a_lds(const _Float16* base, int ld,
                                           int mBase, int kBase, int mL, int half) {
    const _Float16* p = base + (mBase + mL) * ld + kBase + half * 8;
    v16h a;
    ((v8h*)&a)[0] = *(const v8h*)p;
    ((v8h*)&a)[1] = *(const v8h*)(p + 16);
    return a;
}

// B-fragment (32x16 f16, K x N) from a row-major (N_out, K) weight matrix W:
// B[k][n] = W[n][k]. lane<16: col n=nBase+lane, K=kb+0..7, kb+16..23; lane>=16: +8 on each half.
__device__ __forceinline__ v16h load_b_w(const _Float16* W, int ldw,
                                         int nBase, int kBase, int mL, int half) {
    const _Float16* p = W + (nBase + mL) * ldw + kBase + half * 8;
    v16h b;
    ((v8h*)&b)[0] = *(const v8h*)p;
    ((v8h*)&b)[1] = *(const v8h*)(p + 16);
    return b;
}

// ---------------------------------------------------------------------------
// Main recurrence step: si|sf GEMM -> h_cur/c gating -> gate GEMM -> LSTM.
// Block owns 32 rows; 8 waves.
// ---------------------------------------------------------------------------
__global__ __launch_bounds__(256) void step_main(
    const float* __restrict__ xin, int t,
    float* __restrict__ h, float* __restrict__ c,
    const float* __restrict__ ha, const float* __restrict__ hb,
    const _Float16* __restrict__ wsisf,   // (256 out, 256 K) f16: [Wsi; Wsf]
    const _Float16* __restrict__ whh16,   // (512 out, 128 K) f16
    const float* __restrict__ bsisf,      // 256: [bsi, bsf]
    const float* __restrict__ bihh,       // 512: b_ih + b_hh
    const float* __restrict__ wih)        // (512, 3) f32
{
    __shared__ __align__(16) _Float16 lsA[MT * SPLD]; // sp, then hcur(0..127)+sf(132..259)
    __shared__ float lsX[MT][4];

    const int tid  = threadIdx.x;
    const int lane = tid & 31;
    const int wv   = tid >> 5;
    const int mL   = lane & 15;
    const int half = lane >> 4;
    const int r0   = blockIdx.x * MT;

    // ---- phase 0: stage sp = [ha | hb] (f32->f16) and x_t into LDS ----
    for (int idx = tid; idx < MT * 64; idx += 256) {   // 64 float4 per 256-wide row
        int m = idx >> 6, q4 = idx & 63;
        const float* src = (q4 < 32) ? &ha[(size_t)(r0 + m) * HID + q4 * 4]
                                     : &hb[(size_t)(r0 + m) * HID + (q4 - 32) * 4];
        v4f v = *(const v4f*)src;
        _Float16* d = &lsA[m * SPLD + q4 * 4];
        d[0] = (_Float16)v.x; d[1] = (_Float16)v.y;
        d[2] = (_Float16)v.z; d[3] = (_Float16)v.w;
    }
    for (int idx = tid; idx < MT; idx += 256) {
        int row = r0 + idx;
        int b = row / SPA, s = row - b * SPA;
        const float* xp = &xin[(((size_t)b * TSTP + t) * SPA + s) * 3];
        lsX[idx][0] = xp[0]; lsX[idx][1] = xp[1]; lsX[idx][2] = xp[2]; lsX[idx][3] = 0.f;
    }
    __syncthreads();

    // ---- phase 1: (32 x 256) @ wsisf^T -> 32 x 256, tiles (mi in 0..1, ni in 0..15) ----
    v8f acc1[4];
    #pragma unroll
    for (int j = 0; j < 4; ++j) {
        int tt = wv + 8 * j;
        int mi = tt & 1, ni = tt >> 1;
        v8f acc = {};
        #pragma unroll
        for (int kk = 0; kk < 8; ++kk) {
            v16h a = load_a_lds(lsA, SPLD, mi * 16, kk * 32, mL, half);
            v16h b = load_b_w(wsisf, 256, ni * 16, kk * 32, mL, half);
            acc = __builtin_amdgcn_wmma_f32_16x16x32_f16(false, a, false, b,
                                                         (short)0, acc, false, false);
        }
        acc1[j] = acc;
    }
    __syncthreads();  // all waves done reading sp from lsA

    // ---- phase 1 epilogue: sigmoid; hcur = h*si -> LDS f16; sf -> LDS f16 ----
    #pragma unroll
    for (int j = 0; j < 4; ++j) {
        int tt = wv + 8 * j;
        int mi = tt & 1, ni = tt >> 1;
        int n  = ni * 16 + mL;                 // 0..255 (si: <128, sf: >=128)
        float bias = bsisf[n];
        #pragma unroll
        for (int r = 0; r < 8; ++r) {
            int m = mi * 16 + half * 8 + r;
            float sg = sigm(acc1[j][r] + bias);
            if (n < HID) {
                float hv = h[(size_t)(r0 + m) * HID + n];
                lsA[m * SPLD + n] = (_Float16)(hv * sg);
            } else {
                lsA[m * SPLD + 132 + (n - HID)] = (_Float16)sg;
            }
        }
    }
    __syncthreads();

    // ---- phase 2: gates = hcur @ W_hh^T; 16 groups (mi in 0..1, nj in 0..7),
    //      each group computes all 4 gate quadrants so i/f/g/o fragments align ----
    #pragma unroll
    for (int j = 0; j < 2; ++j) {
        int g  = wv + 8 * j;
        int mi = g & 1, nj = g >> 1;
        v8f acc[4] = {{}, {}, {}, {}};
        #pragma unroll
        for (int kk = 0; kk < 4; ++kk) {
            v16h a = load_a_lds(lsA, SPLD, mi * 16, kk * 32, mL, half);
            #pragma unroll
            for (int q = 0; q < 4; ++q) {
                v16h b = load_b_w(whh16, HID, q * HID + nj * 16, kk * 32, mL, half);
                acc[q] = __builtin_amdgcn_wmma_f32_16x16x32_f16(false, a, false, b,
                                                                (short)0, acc[q], false, false);
            }
        }
        int nl = nj * 16 + mL;  // gate column 0..127
        float bb[4], wi[4][3];
        #pragma unroll
        for (int q = 0; q < 4; ++q) {
            bb[q]    = bihh[q * HID + nl];
            wi[q][0] = wih[(q * HID + nl) * 3 + 0];
            wi[q][1] = wih[(q * HID + nl) * 3 + 1];
            wi[q][2] = wih[(q * HID + nl) * 3 + 2];
        }
        #pragma unroll
        for (int r = 0; r < 8; ++r) {
            int m = mi * 16 + half * 8 + r;
            size_t row = (size_t)(r0 + m) * HID + nl;
            float x0 = lsX[m][0], x1 = lsX[m][1], x2 = lsX[m][2];
            float g4[4];
            #pragma unroll
            for (int q = 0; q < 4; ++q)
                g4[q] = acc[q][r] + bb[q] + wi[q][0] * x0 + wi[q][1] * x1 + wi[q][2] * x2;
            float sfv  = (float)lsA[m * SPLD + 132 + nl];
            float ccur = c[row] * sfv;
            float cn   = sigm(g4[1]) * ccur + sigm(g4[0]) * tanhf(g4[2]);
            float hn   = sigm(g4[3]) * tanhf(cn);
            c[row] = cn;
            h[row] = hn;
        }
    }
}

// ---------------------------------------------------------------------------
// Spatial scatter: ha[:,before[i]] += h[:,after[i]]; hb[:,after[i]] += h[:,before[i]]
// i-th index not in {e0<e1<e2<e3} via skip trick.
// ---------------------------------------------------------------------------
__device__ __forceinline__ int skipidx(int i, int e0, int e1, int e2, int e3) {
    int x = i;
    if (x >= e0) x++;
    if (x >= e1) x++;
    if (x >= e2) x++;
    if (x >= e3) x++;
    return x;
}

__global__ void spatial_add_k(const float* __restrict__ h,
                              float* __restrict__ ha, float* __restrict__ hb) {
    int idx = blockIdx.x * blockDim.x + threadIdx.x;   // BAT*96*HID
    if (idx >= BAT * 96 * HID) return;
    int hcol = idx & (HID - 1);
    int rest = idx >> 7;
    int b = rest / 96, i = rest - b * 96;
    int af = skipidx(i, 0, JB1 + 1, JB4, JB6);          // not_after  = {0,31,70,99}
    int bf = skipidx(i, JB1, JB3, JB6 - 1, JB6);        // not_before = {30,60,98,99}
    size_t base = (size_t)b * SPA * HID;
    ha[base + (size_t)bf * HID + hcol] += h[base + (size_t)af * HID + hcol];
    hb[base + (size_t)af * HID + hcol] += h[base + (size_t)bf * HID + hcol];
}

// ---------------------------------------------------------------------------
// Junction overwrites (small matmuls with W_end / W_inter + one copy).
// ---------------------------------------------------------------------------
__global__ void junction_k(const float* __restrict__ h,
                           float* __restrict__ ha, float* __restrict__ hb,
                           const float* __restrict__ Wend, const float* __restrict__ bend,
                           const float* __restrict__ Winter, const float* __restrict__ binter) {
    int idx = blockIdx.x * blockDim.x + threadIdx.x;   // BAT*HID*5
    if (idx >= BAT * HID * 5) return;
    int task = idx / (BAT * HID);
    int rem  = idx - task * (BAT * HID);
    int b = rem >> 7, ho = rem & (HID - 1);
    const float* hrow = &h[(size_t)b * SPA * HID];
    if (task == 4) {            // ha[:,99] = h[:,70]
        ha[((size_t)b * SPA + JB6) * HID + ho] = hrow[JB4 * HID + ho];
        return;
    }
    if (task == 3) {            // hb[:,99] = [h30,h60] @ W_inter^T + b_inter
        float acc = binter[ho];
        const float* w = &Winter[(size_t)ho * 256];
        for (int k = 0; k < HID; ++k) acc += w[k] * hrow[JB1 * HID + k];
        for (int k = 0; k < HID; ++k) acc += w[HID + k] * hrow[JB3 * HID + k];
        hb[((size_t)b * SPA + JB6) * HID + ho] = acc;
        return;
    }
    int s0, s1, s2, ds; float* dst;
    if (task == 0)      { s0 = JB3; s1 = JB4; s2 = JB6; dst = ha; ds = JB1; } // maj -> ha[30]
    else if (task == 1) { s0 = JB1; s1 = JB4; s2 = JB6; dst = ha; ds = JB3; } // min -> ha[60]
    else                { s0 = JB1; s1 = JB3; s2 = JB6; dst = hb; ds = JB4; } // endb -> hb[70]
    float acc = bend[ho];
    const float* w = &Wend[(size_t)ho * 384];
    for (int k = 0; k < HID; ++k) acc += w[k]           * hrow[s0 * HID + k];
    for (int k = 0; k < HID; ++k) acc += w[HID + k]     * hrow[s1 * HID + k];
    for (int k = 0; k < HID; ++k) acc += w[2 * HID + k] * hrow[s2 * HID + k];
    dst[((size_t)b * SPA + ds) * HID + ho] = acc;
}

// ---------------------------------------------------------------------------
// Output head for one timestep: out[:,u,:] = W_out2 @ relu(W_out1 @ h + b1) + b2.
// WMMA for stage 1; deterministic LDS reduction for the final dot.
// ---------------------------------------------------------------------------
__global__ __launch_bounds__(256) void head_k(
    const float* __restrict__ h,
    const _Float16* __restrict__ wout1,   // (128,128) f16
    const float* __restrict__ bout1, const float* __restrict__ wout2,
    const float* __restrict__ bout2, float* __restrict__ out, int u) {
    __shared__ __align__(16) _Float16 lsH[MT * HLD];
    __shared__ float lsP[MT][HID];
    const int tid = threadIdx.x, lane = tid & 31, wv = tid >> 5;
    const int mL = lane & 15, half = lane >> 4;
    const int r0 = blockIdx.x * MT;

    for (int idx = tid; idx < MT * 32; idx += 256) {   // 32 float4 per row
        int m = idx >> 5, q4 = idx & 31;
        v4f v = *(const v4f*)&h[(size_t)(r0 + m) * HID + q4 * 4];
        _Float16* d = &lsH[m * HLD + q4 * 4];
        d[0] = (_Float16)v.x; d[1] = (_Float16)v.y;
        d[2] = (_Float16)v.z; d[3] = (_Float16)v.w;
    }
    __syncthreads();

    #pragma unroll
    for (int j = 0; j < 2; ++j) {
        int tt = wv + 8 * j;
        int mi = tt & 1, nj = tt >> 1;
        v8f acc = {};
        #pragma unroll
        for (int kk = 0; kk < 4; ++kk) {
            v16h a = load_a_lds(lsH, HLD, mi * 16, kk * 32, mL, half);
            v16h b = load_b_w(wout1, HID, nj * 16, kk * 32, mL, half);
            acc = __builtin_amdgcn_wmma_f32_16x16x32_f16(false, a, false, b,
                                                         (short)0, acc, false, false);
        }
        int n = nj * 16 + mL;
        float b1v = bout1[n], w2 = wout2[n];
        #pragma unroll
        for (int r = 0; r < 8; ++r) {
            int m = mi * 16 + half * 8 + r;
            float v = acc[r] + b1v;
            v = v > 0.f ? v : 0.f;
            lsP[m][n] = v * w2;
        }
    }
    __syncthreads();

    if (tid < MT) {
        int row = r0 + tid;
        float s = bout2[0];
        for (int n = 0; n < HID; ++n) s += lsP[tid][n];
        int b = row / SPA, sp = row - b * SPA;
        out[((size_t)b * TOUT + u) * SPA + sp] = s;
    }
}

// ---------------------------------------------------------------------------
// Prep: weight conversion to f16 + fused biases (re-run every call; cheap).
// ---------------------------------------------------------------------------
__global__ void prep_k(const float* __restrict__ Wsi, const float* __restrict__ Wsf,
                       const float* __restrict__ bsi, const float* __restrict__ bsf,
                       const float* __restrict__ Whh, const float* __restrict__ bih,
                       const float* __restrict__ bhh, const float* __restrict__ Wout1,
                       _Float16* __restrict__ wsisf, _Float16* __restrict__ whh16,
                       _Float16* __restrict__ wout116, float* __restrict__ bsisf,
                       float* __restrict__ bihh) {
    int idx = blockIdx.x * blockDim.x + threadIdx.x;
    if (idx < 65536) {  // [Wsi; Wsf] -> (256 out, 256 K)
        int n = idx >> 8, k = idx & 255;
        wsisf[idx] = (_Float16)(n < HID ? Wsi[n * 256 + k] : Wsf[(n - HID) * 256 + k]);
        return;
    }
    idx -= 65536;
    if (idx < 65536) { whh16[idx] = (_Float16)Whh[idx]; return; }
    idx -= 65536;
    if (idx < 16384) { wout116[idx] = (_Float16)Wout1[idx]; return; }
    idx -= 16384;
    if (idx < 256) { bsisf[idx] = (idx < HID) ? bsi[idx] : bsf[idx - HID]; return; }
    idx -= 256;
    if (idx < 512) { bihh[idx] = bih[idx] + bhh[idx]; return; }
}

__global__ void zero_k(float* __restrict__ p, int n4) {
    int i = blockIdx.x * blockDim.x + threadIdx.x;
    if (i < n4) {
        v4f z = {0.f, 0.f, 0.f, 0.f};
        ((v4f*)p)[i] = z;
    }
}

__global__ void const0_k(const float* __restrict__ bout1, const float* __restrict__ wout2,
                         const float* __restrict__ bout2, float* __restrict__ out) {
    int idx = blockIdx.x * blockDim.x + threadIdx.x;   // ROWS
    if (idx >= ROWS) return;
    float s = bout2[0];
    for (int n = 0; n < HID; ++n) {
        float v = bout1[n];
        v = v > 0.f ? v : 0.f;
        s += v * wout2[n];
    }
    int b = idx / SPA, sp = idx - b * SPA;
    out[((size_t)b * TOUT + 0) * SPA + sp] = s;
}

// ---------------------------------------------------------------------------
extern "C" void kernel_launch(void* const* d_in, const int* in_sizes, int n_in,
                              void* d_out, int out_size, void* d_ws, size_t ws_size,
                              hipStream_t stream) {
    (void)in_sizes; (void)n_in; (void)out_size;
    const float* xin    = (const float*)d_in[0];
    // d_in[1] = bucket_number (constants baked in)
    const float* Wih    = (const float*)d_in[2];
    const float* Whh    = (const float*)d_in[3];
    const float* bih    = (const float*)d_in[4];
    const float* bhh    = (const float*)d_in[5];
    const float* Wsf    = (const float*)d_in[6];
    const float* bsf    = (const float*)d_in[7];
    const float* Wsi    = (const float*)d_in[8];
    const float* bsi    = (const float*)d_in[9];
    const float* Wend   = (const float*)d_in[10];
    const float* bend   = (const float*)d_in[11];
    const float* Winter = (const float*)d_in[12];
    const float* binter = (const float*)d_in[13];
    const float* Wout1  = (const float*)d_in[14];
    const float* bout1  = (const float*)d_in[15];
    const float* Wout2  = (const float*)d_in[16];
    const float* bout2  = (const float*)d_in[17];
    float* out = (float*)d_out;

    // Workspace layout (~13.4 MB): h, c, ha, hb (f32) then f16 weights + biases.
    const size_t RH = (size_t)ROWS * HID;
    float* hbuf  = (float*)d_ws;
    float* cbuf  = hbuf + RH;
    float* habuf = cbuf + RH;
    float* hbbuf = habuf + RH;
    char* p = (char*)(hbbuf + RH);
    _Float16* wsisf   = (_Float16*)p; p += 65536 * 2;
    _Float16* whh16   = (_Float16*)p; p += 65536 * 2;
    _Float16* wout116 = (_Float16*)p; p += 16384 * 2;
    float* bsisf = (float*)p; p += 256 * 4;
    float* bihh  = (float*)p; p += 512 * 4;
    (void)ws_size;

    const int prepN = 65536 + 65536 + 16384 + 256 + 512;
    prep_k<<<(prepN + 255) / 256, 256, 0, stream>>>(Wsi, Wsf, bsi, bsf, Whh, bih, bhh,
                                                    Wout1, wsisf, whh16, wout116, bsisf, bihh);
    const int n4 = (int)(4 * RH / 4);
    zero_k<<<(n4 + 255) / 256, 256, 0, stream>>>(hbuf, n4);
    const0_k<<<(ROWS + 255) / 256, 256, 0, stream>>>(bout1, Wout2, bout2, out);

    for (int t = 0; t < TSTP; ++t) {
        step_main<<<ROWS / MT, 256, 0, stream>>>(xin, t, hbuf, cbuf, habuf, hbbuf,
                                                 wsisf, whh16, bsisf, bihh, Wih);
        spatial_add_k<<<(BAT * 96 * HID + 255) / 256, 256, 0, stream>>>(hbuf, habuf, hbbuf);
        junction_k<<<(BAT * HID * 5 + 255) / 256, 256, 0, stream>>>(hbuf, habuf, hbbuf,
                                                                    Wend, bend, Winter, binter);
        if (t >= TPRED + 1)
            head_k<<<ROWS / MT, 256, 0, stream>>>(hbuf, wout116, bout1, Wout2, bout2,
                                                  out, t - TPRED);
    }
}
